// DeformableTransformerDecoderLayer_58789512347735
// MI455X (gfx1250) — matrix operand order repaired and checked by reading, not compiled, not run on hardware
//
#include <hip/hip_runtime.h>
#include <math.h>

// ---------------------------------------------------------------------------
// Deformable Transformer decoder layer for MI455X (gfx1250, wave32, WMMA).
// GEMMs use v_wmma_f32_16x16x32_f16 (f32 accumulate). Weights are pre-packed
// into per-lane fragment order so B loads are one contiguous 32B load/lane.
// K-loop is software-pipelined: next step's A+B fragments are fetched while
// the current step's 4 WMMAs execute, hiding load latency inside each wave.
// ---------------------------------------------------------------------------

typedef _Float16 v16h __attribute__((ext_vector_type(16)));
typedef float    v8f  __attribute__((ext_vector_type(8)));
typedef float    v4f  __attribute__((ext_vector_type(4)));

constexpr int  B_   = 8;
constexpr int  LQ_  = 300;
constexpr int  NH_  = 8;
constexpr int  LV_  = 13294;            // 100^2 + 50^2 + 25^2 + 13^2
constexpr int  NQ_  = B_ * LQ_;         // 2400
constexpr long MV_  = (long)B_ * LV_;   // 106352 rows for value projection

// ---------------------------------------------------------------------------
// Pack f32 weight [K][N] into WMMA B-fragment order (f16):
// tile = kt*(N/16)+nt ; lane holds column n = nt*16+(lane&15),
// 16 consecutive K values starting at kt*32 + (lane>>4)*16  (contiguous k order, same as A).
// ---------------------------------------------------------------------------
__global__ void pack_w(const float* __restrict__ W, _Float16* __restrict__ Wp,
                       int K, int N) {
  int t = blockIdx.x * blockDim.x + threadIdx.x;
  int total = (K >> 5) * (N >> 4) * 32;
  if (t >= total) return;
  int lane = t & 31;
  int tile = t >> 5;
  int ntiles = N >> 4;
  int kt = tile / ntiles;
  int nt = tile - kt * ntiles;
  int n  = nt * 16 + (lane & 15);
  int k0 = kt * 32 + (lane >> 4) * 16;
  v16h o;
#pragma unroll
  for (int j = 0; j < 16; ++j) o[j] = (_Float16)W[(size_t)(k0 + j) * N + n];
  *(v16h*)(Wp + (size_t)t * 16) = o;
}

// ---------------------------------------------------------------------------
// Generic WMMA GEMM:  out[M][N] = A[M][K] (f32, converted to f16) * Wp + bias
// One wave -> 16 rows x 64 cols (4 C tiles).  EPI: 0 = f32 out,
// 1 = ReLU f32 out, 2 = f16 out with per-row padding mask (value projection).
// ---------------------------------------------------------------------------
template <int K, int N, int EPI>
__global__ __launch_bounds__(256) void gemm_wmma(
    const float* __restrict__ A, const _Float16* __restrict__ Wp,
    const float* __restrict__ bias, float* __restrict__ outF,
    _Float16* __restrict__ outH, const unsigned char* __restrict__ rowmask,
    int M) {
  constexpr int NB = N / 64;
  constexpr int KT = K / 32;
  constexpr int WSTRIDE = (N / 16) * 512;  // halves per K-step in packed W
  int wave = (blockIdx.x * blockDim.x + threadIdx.x) >> 5;
  int lane = threadIdx.x & 31;
  int mt = wave / NB;
  int nb = wave - mt * NB;
  if (mt * 16 >= M) return;
  int hs = lane >> 4;
  int lm = lane & 15;
  // A fragment base: contiguous k order (lanes<16: K 0..15; hi half K 16..31)
  const float* arow = A + (size_t)(mt * 16 + lm) * K + hs * 16;
  const _Float16* wbase = Wp + ((size_t)(nb * 4) * 32 + lane) * 16;

  auto loadA = [&](int kt) -> v16h {
    const float* ap = arow + kt * 32;
    v4f a0 = *(const v4f*)(ap);
    v4f a1 = *(const v4f*)(ap + 4);
    v4f a2 = *(const v4f*)(ap + 8);
    v4f a3 = *(const v4f*)(ap + 12);
    v16h af;
    af[0]  = (_Float16)a0[0]; af[1]  = (_Float16)a0[1];
    af[2]  = (_Float16)a0[2]; af[3]  = (_Float16)a0[3];
    af[4]  = (_Float16)a1[0]; af[5]  = (_Float16)a1[1];
    af[6]  = (_Float16)a1[2]; af[7]  = (_Float16)a1[3];
    af[8]  = (_Float16)a2[0]; af[9]  = (_Float16)a2[1];
    af[10] = (_Float16)a2[2]; af[11] = (_Float16)a2[3];
    af[12] = (_Float16)a3[0]; af[13] = (_Float16)a3[1];
    af[14] = (_Float16)a3[2]; af[15] = (_Float16)a3[3];
    return af;
  };

  v8f c0 = {}, c1 = {}, c2 = {}, c3 = {};
  // Prologue: fragments for K-step 0.
  v16h af = loadA(0);
  v16h b0 = *(const v16h*)(wbase);
  v16h b1 = *(const v16h*)(wbase + 512);
  v16h b2 = *(const v16h*)(wbase + 1024);
  v16h b3 = *(const v16h*)(wbase + 1536);

  for (int kt = 0; kt < KT - 1; ++kt) {
    // Prefetch next K-step while current WMMAs run.
    v16h afn = loadA(kt + 1);
    const _Float16* wn = wbase + (size_t)(kt + 1) * WSTRIDE;
    v16h b0n = *(const v16h*)(wn);
    v16h b1n = *(const v16h*)(wn + 512);
    v16h b2n = *(const v16h*)(wn + 1024);
    v16h b3n = *(const v16h*)(wn + 1536);
    c0 = __builtin_amdgcn_wmma_f32_16x16x32_f16(false, af, false, b0, (short)0, c0, false, false);
    c1 = __builtin_amdgcn_wmma_f32_16x16x32_f16(false, af, false, b1, (short)0, c1, false, false);
    c2 = __builtin_amdgcn_wmma_f32_16x16x32_f16(false, af, false, b2, (short)0, c2, false, false);
    c3 = __builtin_amdgcn_wmma_f32_16x16x32_f16(false, af, false, b3, (short)0, c3, false, false);
    af = afn; b0 = b0n; b1 = b1n; b2 = b2n; b3 = b3n;
  }
  c0 = __builtin_amdgcn_wmma_f32_16x16x32_f16(false, af, false, b0, (short)0, c0, false, false);
  c1 = __builtin_amdgcn_wmma_f32_16x16x32_f16(false, af, false, b1, (short)0, c1, false, false);
  c2 = __builtin_amdgcn_wmma_f32_16x16x32_f16(false, af, false, b2, (short)0, c2, false, false);
  c3 = __builtin_amdgcn_wmma_f32_16x16x32_f16(false, af, false, b3, (short)0, c3, false, false);

  v8f cs[4] = {c0, c1, c2, c3};
  int colbase = nb * 64 + lm;
#pragma unroll
  for (int j = 0; j < 4; ++j) {
    int col = colbase + j * 16;
    float bc = bias[col];
#pragma unroll
    for (int v = 0; v < 8; ++v) {
      int r = mt * 16 + hs * 8 + v;   // C layout: vgpr v -> row v (+8 for hi half)
      float val = cs[j][v] + bc;
      if (EPI == 1) val = fmaxf(val, 0.f);
      if (EPI == 2) {
        if (rowmask[r]) val = 0.f;
        outH[(size_t)r * N + col] = (_Float16)val;
      } else {
        outF[(size_t)r * N + col] = val;
      }
    }
  }
}

// ---------------------------------------------------------------------------
__global__ void add_vec(const float* __restrict__ a, const float* __restrict__ b,
                        float* __restrict__ o, int n) {
  int i = blockIdx.x * blockDim.x + threadIdx.x;
  if (i < n) o[i] = a[i] + b[i];
}

// out = LayerNorm(A + Bv) * g + be   (one 256-thread block per row, D=256)
__global__ __launch_bounds__(256) void add_ln(
    const float* __restrict__ A, const float* __restrict__ Bv,
    const float* __restrict__ g, const float* __restrict__ be,
    float* __restrict__ out) {
  int row = blockIdx.x, t = threadIdx.x;
  __shared__ float red[256];
  float v = A[(size_t)row * 256 + t] + Bv[(size_t)row * 256 + t];
  red[t] = v;
  __syncthreads();
  for (int s = 128; s > 0; s >>= 1) {
    if (t < s) red[t] += red[t + s];
    __syncthreads();
  }
  float mean = red[0] * (1.f / 256.f);
  __syncthreads();
  float d = v - mean;
  red[t] = d * d;
  __syncthreads();
  for (int s = 128; s > 0; s >>= 1) {
    if (t < s) red[t] += red[t + s];
    __syncthreads();
  }
  float var = red[0] * (1.f / 256.f);
  out[(size_t)row * 256 + t] = d * rsqrtf(var + 1e-5f) * g[t] + be[t];
}

// ---------------------------------------------------------------------------
// MS-Deformable-Attention sampling: one wave per (b, q, head); lane = channel.
// Softmax over 16 (level,point) logits, 4-corner bilinear gather on f16 value.
// ---------------------------------------------------------------------------
__global__ __launch_bounds__(256) void msda_sample(
    const _Float16* __restrict__ valH,  // [B*LV][256], col = h*32 + d
    const float* __restrict__ offb,     // [NQ][256]
    const float* __restrict__ alog,     // [NQ][128]
    const float* __restrict__ refp,     // [B][LQ][4][2]
    float* __restrict__ acc) {          // [NQ][256]
  int wave = (blockIdx.x * blockDim.x + threadIdx.x) >> 5;
  int lane = threadIdx.x & 31;
  if (wave >= NQ_ * NH_) return;
  int h  = wave & 7;
  int bq = wave >> 3;
  int b  = bq / LQ_;

  const float* lg = alog + (size_t)bq * 128 + h * 16;
  float w[16];
  float mx = lg[0];
#pragma unroll
  for (int i = 1; i < 16; ++i) mx = fmaxf(mx, lg[i]);
  float s = 0.f;
#pragma unroll
  for (int i = 0; i < 16; ++i) { w[i] = __expf(lg[i] - mx); s += w[i]; }
  float inv = 1.f / s;

  const float* ofp = offb + (size_t)bq * 256 + h * 32;
  const int startL[4] = {0, 10000, 12500, 13125};
  const int HL[4] = {100, 50, 25, 13};
  float a = 0.f;
#pragma unroll
  for (int l = 0; l < 4; ++l) {
    int H = HL[l], W = HL[l];  // square levels
    float fW = (float)W, fH = (float)H;
    float rx = refp[((size_t)bq * 4 + l) * 2 + 0];
    float ry = refp[((size_t)bq * 4 + l) * 2 + 1];
    const _Float16* vb =
        valH + ((size_t)b * LV_ + startL[l]) * 256 + h * 32 + lane;
#pragma unroll
    for (int p = 0; p < 4; ++p) {
      float ix = (rx + ofp[l * 8 + p * 2 + 0] / fW) * fW - 0.5f;
      float iy = (ry + ofp[l * 8 + p * 2 + 1] / fH) * fH - 0.5f;
      float fx0 = floorf(ix), fy0 = floorf(iy);
      float wx = ix - fx0, wy = iy - fy0;
      int x0 = (int)fx0, y0 = (int)fy0;
      int x1 = x0 + 1, y1 = y0 + 1;
      float smp = 0.f;
      if (x0 >= 0 && x0 < W && y0 >= 0 && y0 < H)
        smp += (1.f - wx) * (1.f - wy) * (float)vb[(size_t)(y0 * W + x0) * 256];
      if (x1 >= 0 && x1 < W && y0 >= 0 && y0 < H)
        smp += wx * (1.f - wy) * (float)vb[(size_t)(y0 * W + x1) * 256];
      if (x0 >= 0 && x0 < W && y1 >= 0 && y1 < H)
        smp += (1.f - wx) * wy * (float)vb[(size_t)(y1 * W + x0) * 256];
      if (x1 >= 0 && x1 < W && y1 >= 0 && y1 < H)
        smp += wx * wy * (float)vb[(size_t)(y1 * W + x1) * 256];
      a += inv * w[l * 4 + p] * smp;
    }
  }
  acc[(size_t)bq * 256 + h * 32 + lane] = a;
}

// ---------------------------------------------------------------------------
extern "C" void kernel_launch(void* const* d_in, const int* in_sizes, int n_in,
                              void* d_out, int out_size, void* d_ws,
                              size_t ws_size, hipStream_t stream) {
  (void)in_sizes; (void)n_in; (void)out_size; (void)ws_size;
  const float* tgt  = (const float*)d_in[0];
  const float* qpos = (const float*)d_in[1];
  const float* refp = (const float*)d_in[2];
  const float* src  = (const float*)d_in[3];
  const unsigned char* mask = (const unsigned char*)d_in[4];
  const float* Wv = (const float*)d_in[5];   const float* bv  = (const float*)d_in[6];
  const float* Wo = (const float*)d_in[7];   const float* bo  = (const float*)d_in[8];
  const float* Wa = (const float*)d_in[9];   const float* ba  = (const float*)d_in[10];
  const float* Wq = (const float*)d_in[11];  const float* bq_ = (const float*)d_in[12];
  const float* g1 = (const float*)d_in[13];  const float* be1 = (const float*)d_in[14];
  const float* W1 = (const float*)d_in[15];  const float* c1  = (const float*)d_in[16];
  const float* W2 = (const float*)d_in[17];  const float* c2  = (const float*)d_in[18];
  const float* g3 = (const float*)d_in[19];  const float* be3 = (const float*)d_in[20];

  char* ws = (char*)d_ws;
  size_t cur = 0;
  auto alloc = [&](size_t bytes) {
    size_t r = cur;
    cur = (cur + bytes + 255) & ~(size_t)255;
    return r;
  };
  _Float16* valH = (_Float16*)(ws + alloc((size_t)MV_ * 256 * 2));
  float* q    = (float*)(ws + alloc((size_t)NQ_ * 256 * 4));
  float* offb = (float*)(ws + alloc((size_t)NQ_ * 256 * 4));
  float* alog = (float*)(ws + alloc((size_t)NQ_ * 128 * 4));
  float* acc  = (float*)(ws + alloc((size_t)NQ_ * 256 * 4));
  float* ao   = (float*)(ws + alloc((size_t)NQ_ * 256 * 4));
  float* x    = (float*)(ws + alloc((size_t)NQ_ * 256 * 4));
  float* h1   = (float*)(ws + alloc((size_t)NQ_ * 1024 * 4));
  float* f2   = (float*)(ws + alloc((size_t)NQ_ * 256 * 4));
  _Float16* WvP = (_Float16*)(ws + alloc(256 * 256 * 2));
  _Float16* WoP = (_Float16*)(ws + alloc(256 * 256 * 2));
  _Float16* WaP = (_Float16*)(ws + alloc(256 * 128 * 2));
  _Float16* WqP = (_Float16*)(ws + alloc(256 * 256 * 2));
  _Float16* W1P = (_Float16*)(ws + alloc(256 * 1024 * 2));
  _Float16* W2P = (_Float16*)(ws + alloc(1024 * 256 * 2));

  const int TB = 256;
  auto cdiv = [](long a, long b) { return (int)((a + b - 1) / b); };

  // Pack weights into WMMA fragment order.
  pack_w<<<cdiv(8 * 16 * 32, TB), TB, 0, stream>>>(Wv, WvP, 256, 256);
  pack_w<<<cdiv(8 * 16 * 32, TB), TB, 0, stream>>>(Wo, WoP, 256, 256);
  pack_w<<<cdiv(8 * 8 * 32, TB), TB, 0, stream>>>(Wa, WaP, 256, 128);
  pack_w<<<cdiv(8 * 16 * 32, TB), TB, 0, stream>>>(Wq, WqP, 256, 256);
  pack_w<<<cdiv(8 * 64 * 32, TB), TB, 0, stream>>>(W1, W1P, 256, 1024);
  pack_w<<<cdiv(32 * 16 * 32, TB), TB, 0, stream>>>(W2, W2P, 1024, 256);

  // q = tgt + query_pos
  add_vec<<<cdiv((long)NQ_ * 256, TB), TB, 0, stream>>>(tgt, qpos, q, NQ_ * 256);

  // value = mask(src @ Wv + bv) -> f16   [106352 x 256]
  gemm_wmma<256, 256, 2><<<cdiv((MV_ / 16) * 4, 8), TB, 0, stream>>>(
      src, WvP, bv, nullptr, valH, mask, (int)MV_);
  // sampling offsets and attention logits
  gemm_wmma<256, 256, 0><<<cdiv((NQ_ / 16) * 4, 8), TB, 0, stream>>>(
      q, WoP, bo, offb, nullptr, nullptr, NQ_);
  gemm_wmma<256, 128, 0><<<cdiv((NQ_ / 16) * 2, 8), TB, 0, stream>>>(
      q, WaP, ba, alog, nullptr, nullptr, NQ_);

  // deformable sampling (softmax + bilinear gather + weighted sum)
  msda_sample<<<cdiv((long)NQ_ * NH_ * 32, TB), TB, 0, stream>>>(
      valH, offb, alog, refp, acc);

  // output projection
  gemm_wmma<256, 256, 0><<<cdiv((NQ_ / 16) * 4, 8), TB, 0, stream>>>(
      acc, WqP, bq_, ao, nullptr, nullptr, NQ_);

  // x = LN(tgt + attn_out)
  add_ln<<<NQ_, 256, 0, stream>>>(tgt, ao, g1, be1, x);

  // FFN: relu(x@W1+c1)@W2+c2
  gemm_wmma<256, 1024, 1><<<cdiv((NQ_ / 16) * 16, 8), TB, 0, stream>>>(
      x, W1P, c1, h1, nullptr, nullptr, NQ_);
  gemm_wmma<1024, 256, 0><<<cdiv((NQ_ / 16) * 4, 8), TB, 0, stream>>>(
      h1, W2P, c2, f2, nullptr, nullptr, NQ_);

  // out = LN(x + ffn)
  add_ln<<<NQ_, 256, 0, stream>>>(x, f2, g3, be3, (float*)d_out);
}
